// QMOIAttention_18794776888072
// MI455X (gfx1250) — matrix operand order repaired and checked by reading, not compiled, not run on hardware
//
#include <hip/hip_runtime.h>
#include <hip/hip_bf16.h>
#include <math.h>

// ---------------------------------------------------------------------------
// Problem constants (B=1)
// ---------------------------------------------------------------------------
#define SDIM 2048
#define EDIM 1024
#define HN   16
#define DDIM 64          // head dim
#define NEGINF -1e30f

typedef unsigned short u16_t;
typedef __attribute__((ext_vector_type(16))) __bf16 v16bf;
typedef __attribute__((ext_vector_type(8)))  float  v8f;

struct U32x8 { uint4 lo, hi; };

// ---------------------------------------------------------------------------
// Helpers
// ---------------------------------------------------------------------------
__device__ __forceinline__ u16_t f2bf(float f) {
    unsigned u = __float_as_uint(f);
    u += 0x7FFFu + ((u >> 16) & 1u);      // round to nearest even
    return (u16_t)(u >> 16);
}

__device__ __forceinline__ v8f zero8() {
    v8f z = {0.f,0.f,0.f,0.f,0.f,0.f,0.f,0.f};
    return z;
}

// 16-bit A fragment (16x32, MxK), wave32:
//   lane l: m = l&15, kb = l>>4
//   elements 0..15 : k = kb*16 + 0..15
__device__ __forceinline__ v16bf load_a_frag(const u16_t* row_k0, int kb) {
    U32x8 t;
    t.lo = *reinterpret_cast<const uint4*>(row_k0 + kb * 16);
    t.hi = *reinterpret_cast<const uint4*>(row_k0 + kb * 16 + 8);
    return __builtin_bit_cast(v16bf, t);
}

// 16-bit B fragment (32x16, KxN), wave32:
//   lane l: n = l&15, kb = l>>4 ; elements = k = kb*16 + 0..15
//   (memory must hold B K-contiguous per n; works for global or LDS pointers)
__device__ __forceinline__ v16bf load_b_frag(const u16_t* row_k0, int kb) {
    U32x8 t;
    const uint4* q = reinterpret_cast<const uint4*>(row_k0 + kb * 16);
    t.lo = q[0];
    t.hi = q[1];
    return __builtin_bit_cast(v16bf, t);
}

__device__ __forceinline__ v8f wmma_bf16(v16bf a, v16bf b, v8f c) {
    return __builtin_amdgcn_wmma_f32_16x16x32_bf16(
        /*neg_a=*/false, a, /*neg_b=*/false, b,
        /*c_mod=*/(short)0, c, /*reuse_a=*/false, /*reuse_b=*/false);
}

// reductions across the 16 lanes holding one matrix row (lane halves)
__device__ __forceinline__ float rowmax16(float v) {
    v = fmaxf(v, __shfl_xor(v, 1, 32));
    v = fmaxf(v, __shfl_xor(v, 2, 32));
    v = fmaxf(v, __shfl_xor(v, 4, 32));
    v = fmaxf(v, __shfl_xor(v, 8, 32));
    return v;
}
__device__ __forceinline__ float rowsum16(float v) {
    v += __shfl_xor(v, 1, 32);
    v += __shfl_xor(v, 2, 32);
    v += __shfl_xor(v, 4, 32);
    v += __shfl_xor(v, 8, 32);
    return v;
}

// --- CDNA5 async global->LDS copy (ASYNCcnt path) ---------------------------
// Copies 16 bytes per lane. LDS destination address is the low 32 bits of the
// generic pointer (aperture truncation per ISA 10.2).
__device__ __forceinline__ void async_copy_b128(void* lds_dst, const void* gsrc) {
    unsigned l = (unsigned)(uintptr_t)lds_dst;
    unsigned long long g = (unsigned long long)(uintptr_t)gsrc;
    asm volatile("global_load_async_to_lds_b128 %0, %1, off"
                 :: "v"(l), "v"(g) : "memory");
}
__device__ __forceinline__ void wait_async16() {
    asm volatile("s_wait_asynccnt 0x10" ::: "memory");
}
__device__ __forceinline__ void wait_async0() {
    asm volatile("s_wait_asynccnt 0x0" ::: "memory");
}

// ---------------------------------------------------------------------------
// Kernel 1: f32 -> bf16 conversion / weight transpose (one-shot prep)
// ---------------------------------------------------------------------------
__global__ void cvt_bf16_kernel(const float* __restrict__ src,
                                u16_t* __restrict__ dst, int n) {
    int i = blockIdx.x * blockDim.x + threadIdx.x;
    if (i < n) dst[i] = f2bf(src[i]);
}

// W[k][n] (ExE) -> WT[n][k] bf16, so B fragments are contiguous in K.
__global__ void transpose_cvt_kernel(const float* __restrict__ src,
                                     u16_t* __restrict__ dst) {
    int i = blockIdx.x * blockDim.x + threadIdx.x;
    if (i < EDIM * EDIM) {
        int k = i >> 10;           // EDIM == 1024
        int n = i & (EDIM - 1);
        dst[(size_t)n * EDIM + k] = f2bf(src[i]);
    }
}

// ---------------------------------------------------------------------------
// Pipelined 16x64-strip GEMM core: acc[4] += A(16xE) x B(Ex64), bf16->f32.
// Explicit ping-pong double buffering (manual 2x unroll) so the register
// rotation needs no v_mov copies. EDIM/32 is even; look-ahead index wraps
// with &(EDIM-1) (the wrapped loads are dead after the loop).
// ---------------------------------------------------------------------------
__device__ __forceinline__ void gemm_strip(const u16_t* __restrict__ arow,
                                           const u16_t* const wrow[4],
                                           int kb, v8f acc[4])
{
    v16bf a0 = load_a_frag(arow, kb);
    v16bf b0_0 = load_b_frag(wrow[0], kb);
    v16bf b0_1 = load_b_frag(wrow[1], kb);
    v16bf b0_2 = load_b_frag(wrow[2], kb);
    v16bf b0_3 = load_b_frag(wrow[3], kb);

    for (int kk = 0; kk < EDIM; kk += 64) {
        const int k1 = kk + 32;                    // always < EDIM
        v16bf a1   = load_a_frag(arow + k1, kb);
        v16bf b1_0 = load_b_frag(wrow[0] + k1, kb);
        v16bf b1_1 = load_b_frag(wrow[1] + k1, kb);
        v16bf b1_2 = load_b_frag(wrow[2] + k1, kb);
        v16bf b1_3 = load_b_frag(wrow[3] + k1, kb);
        acc[0] = wmma_bf16(a0, b0_0, acc[0]);
        acc[1] = wmma_bf16(a0, b0_1, acc[1]);
        acc[2] = wmma_bf16(a0, b0_2, acc[2]);
        acc[3] = wmma_bf16(a0, b0_3, acc[3]);

        const int k2 = (kk + 64) & (EDIM - 1);     // wraps to 0 on last iter
        a0   = load_a_frag(arow + k2, kb);
        b0_0 = load_b_frag(wrow[0] + k2, kb);
        b0_1 = load_b_frag(wrow[1] + k2, kb);
        b0_2 = load_b_frag(wrow[2] + k2, kb);
        b0_3 = load_b_frag(wrow[3] + k2, kb);
        acc[0] = wmma_bf16(a1, b1_0, acc[0]);
        acc[1] = wmma_bf16(a1, b1_1, acc[1]);
        acc[2] = wmma_bf16(a1, b1_2, acc[2]);
        acc[3] = wmma_bf16(a1, b1_3, acc[3]);
    }
}

// ---------------------------------------------------------------------------
// Kernel 2: fused QKV projection + bias + rotary (+ K pre-scaled by D^-1/2)
//   grid (S/64, H, 3), block 128 (4 waves, one 16-row i-tile each)
// ---------------------------------------------------------------------------
__global__ void __launch_bounds__(128)
qkv_kernel(const u16_t* __restrict__ xb,
           const u16_t* __restrict__ wqt, const u16_t* __restrict__ wkt,
           const u16_t* __restrict__ wvt,
           const float* __restrict__ bq, const float* __restrict__ bk,
           const float* __restrict__ bv,
           u16_t* __restrict__ qh, u16_t* __restrict__ kh,
           u16_t* __restrict__ vt)
{
    const int lane = threadIdx.x & 31;
    const int wv   = threadIdx.x >> 5;
    const int it   = blockIdx.x * 4 + wv;
    const int h    = blockIdx.y;
    const int proj = blockIdx.z;
    const int i0   = it * 16;
    const int n0   = h * 64;
    const int m    = lane & 15;
    const int kb   = lane >> 4;

    const u16_t* WT   = (proj == 0) ? wqt : (proj == 1) ? wkt : wvt;
    const float* bias = (proj == 0) ? bq  : (proj == 1) ? bk  : bv;

    const u16_t* xrow = xb + (size_t)(i0 + m) * EDIM;
    const u16_t* wrow[4];
#pragma unroll
    for (int t = 0; t < 4; ++t)
        wrow[t] = WT + (size_t)(n0 + t * 16 + m) * EDIM;

    v8f acc[4];
#pragma unroll
    for (int t = 0; t < 4; ++t) acc[t] = zero8();

    gemm_strip(xrow, wrow, kb, acc);

    // bias (column n0 + t*16 + m, identical across the 8 rows per lane)
#pragma unroll
    for (int t = 0; t < 4; ++t) {
        const float bcol = bias[n0 + t * 16 + m];
#pragma unroll
        for (int r = 0; r < 8; ++r) acc[t][r] += bcol;
    }

    if (proj < 2) {
        // rotary: pair (d, d+32); tiles t and t+2 hold the two halves
        u16_t* dst = (proj == 0) ? qh : kh;
        const float sc = (proj == 1) ? 0.125f : 1.0f;   // fold QK scale into K
#pragma unroll
        for (int t = 0; t < 2; ++t) {
            const int d = t * 16 + m;                   // 0..31
            const float invf = __powf(10000.0f, -(float)d * (1.0f / 32.0f));
#pragma unroll
            for (int r = 0; r < 8; ++r) {
                const int s = i0 + r + 8 * kb;
                float sn, c;
                __sincosf((float)s * invf, &sn, &c);
                const float lo = acc[t][r], hi = acc[t + 2][r];
                const float nlo = (lo * c - hi * sn) * sc;
                const float nhi = (hi * c + lo * sn) * sc;
                u16_t* drow = dst + ((size_t)(h * SDIM + s)) * DDIM;
                drow[d]      = f2bf(nlo);
                drow[d + 32] = f2bf(nhi);
            }
        }
    } else {
        // V transposed: vt[h][d][s]; rows r..r+7 contiguous in s -> b128 store
#pragma unroll
        for (int t = 0; t < 4; ++t) {
            const int d = t * 16 + m;
            u16_t tmp[8];
#pragma unroll
            for (int r = 0; r < 8; ++r) tmp[r] = f2bf(acc[t][r]);
            *reinterpret_cast<uint4*>(
                vt + ((size_t)(h * DDIM + d)) * SDIM + i0 + 8 * kb) =
                *reinterpret_cast<const uint4*>(tmp);
        }
    }
}

// ---------------------------------------------------------------------------
// Kernel 3: flash attention with rotary-Q/K + relative-position bias.
//   grid (S/16, H), block 32 (one wave per 16-row query tile).
//   K/V tiles are staged into LDS with global_load_async_to_lds_b128,
//   double-buffered one j-step ahead (ASYNCcnt pipeline); WMMA B-fragments
//   for QK^T and PV come from LDS (ds_load_b128).
// ---------------------------------------------------------------------------
__global__ void __launch_bounds__(32)
attn_kernel(const u16_t* __restrict__ qh, const u16_t* __restrict__ kh,
            const u16_t* __restrict__ vt, const u16_t* __restrict__ relb,
            u16_t* __restrict__ ao)
{
    __shared__ __align__(16) u16_t kbuf[2][32][DDIM];   // 2 x 4KB
    __shared__ __align__(16) u16_t vbuf[2][DDIM][32];   // 2 x 4KB
    __shared__ __align__(16) float Pbuf[16][48];        // 3KB
    __shared__ __align__(16) u16_t pstage[16][32];      // 1KB

    const int lane = threadIdx.x & 31;
    const int it = blockIdx.x, h = blockIdx.y;
    const int i0 = it * 16;
    const int m  = lane & 15;
    const int kb = lane >> 4;

    // issue async copies of the K tile (contiguous 4KB) and V tile
    // (64 rows x 64B from vt[h][d][s]) for j-step base jj -> buffer b.
    auto issue_tiles = [&](int b, int jj) {
        const char* kg = (const char*)(kh + (size_t)(h * SDIM + jj) * DDIM);
        char* kl = (char*)&kbuf[b][0][0];
#pragma unroll
        for (int i = 0; i < 8; ++i) {
            const int off = (i * 32 + lane) * 16;
            async_copy_b128(kl + off, kg + off);
        }
#pragma unroll
        for (int i = 0; i < 8; ++i) {
            const int row = i * 8 + (lane >> 2);
            const int cs  = (lane & 3) * 8;               // u16 elements
            async_copy_b128(&vbuf[b][row][cs],
                            vt + ((size_t)(h * DDIM + row)) * SDIM + jj + cs);
        }
    };

    const u16_t* qrow = qh + (size_t)(h * SDIM + i0 + m) * DDIM;
    const v16bf qa0 = load_a_frag(qrow, kb);        // K-dim d = 0..31
    const v16bf qa1 = load_a_frag(qrow + 32, kb);   // K-dim d = 32..63

    v8f o[4];
#pragma unroll
    for (int t = 0; t < 4; ++t) o[t] = zero8();
    float mst[8], lst[8];
#pragma unroll
    for (int r = 0; r < 8; ++r) { mst[r] = NEGINF; lst[r] = 0.f; }

    const int nsteps = (i0 + 15) / 32 + 1;          // causal j range
    issue_tiles(0, 0);                              // prologue: step 0 tiles

    int buf = 0;
    for (int stp = 0; stp < nsteps; ++stp) {
        const int jj0 = stp * 32;
        const bool havenext = (stp + 1 < nsteps);
        if (havenext) issue_tiles(buf ^ 1, jj0 + 32);   // overlap with P WMMAs

        // ---- relative-position band P: buckets bb .. bb+47 (global B) ----
        const int bb = jj0 - i0 + (SDIM - 16);      // = jj0-i0+S-1-15 >= 0
#pragma unroll
        for (int pt = 0; pt < 3; ++pt) {
            const u16_t* rr = relb + (size_t)(bb + pt * 16 + m) * DDIM;
            v8f p = zero8();
            p = wmma_bf16(qa0, load_b_frag(rr,      kb), p);
            p = wmma_bf16(qa1, load_b_frag(rr + 32, kb), p);
#pragma unroll
            for (int r = 0; r < 8; ++r)
                Pbuf[r + 8 * kb][pt * 16 + m] = p[r];
        }
        if (havenext)
            __builtin_prefetch(relb + (size_t)(bb + 48 + m) * DDIM, 0, 0);

        // ---- this step's K/V tiles have landed in LDS ----
        if (havenext) wait_async16();   // only next step's 16 copies in flight
        else          wait_async0();

        // ---- scores = (q * scale_in_K) . K^T  (B-frags from LDS) ----
        v8f s0 = zero8(), s1 = zero8();
        s0 = wmma_bf16(qa0, load_b_frag(&kbuf[buf][m][0],       kb), s0);
        s0 = wmma_bf16(qa1, load_b_frag(&kbuf[buf][m][32],      kb), s0);
        s1 = wmma_bf16(qa0, load_b_frag(&kbuf[buf][16 + m][0],  kb), s1);
        s1 = wmma_bf16(qa1, load_b_frag(&kbuf[buf][16 + m][32], kb), s1);

        asm volatile("s_wait_dscnt 0" ::: "memory");  // Pbuf RAW, same wave

        // ---- add rel-pos bias, causal mask, online softmax ----
        const bool lastp = (stp == nsteps - 1);
#pragma unroll
        for (int r = 0; r < 8; ++r) {
            const int mi = r + 8 * kb;
            float a0 = s0[r] + Pbuf[mi][15 + m - mi];
            float a1 = s1[r] + Pbuf[mi][31 + m - mi];
            if (lastp) {
                const int ia = i0 + mi;
                if (jj0 + m > ia)      a0 = NEGINF;
                if (jj0 + 16 + m > ia) a1 = NEGINF;
            }
            float rm = rowmax16(fmaxf(a0, a1));
            const float mnew  = fmaxf(mst[r], rm);
            const float alpha = __expf(mst[r] - mnew);
            const float p0 = __expf(a0 - mnew);
            const float p1 = __expf(a1 - mnew);
            const float rs = rowsum16(p0 + p1);
            lst[r] = lst[r] * alpha + rs;
            mst[r] = mnew;
#pragma unroll
            for (int t = 0; t < 4; ++t) o[t][r] *= alpha;
            pstage[mi][m]      = f2bf(p0);
            pstage[mi][m + 16] = f2bf(p1);
        }
        asm volatile("s_wait_dscnt 0" ::: "memory");

        // ---- o += P . V   (A-frag from LDS pstage, B-frags from LDS vbuf) --
        const v16bf pa = load_a_frag(&pstage[m][0], kb);
#pragma unroll
        for (int t = 0; t < 4; ++t)
            o[t] = wmma_bf16(pa, load_b_frag(&vbuf[buf][t * 16 + m][0], kb),
                             o[t]);
        buf ^= 1;
    }

    // ---- normalize and emit bf16 attention output ao[s][h*64+d] ----
#pragma unroll
    for (int t = 0; t < 4; ++t) {
#pragma unroll
        for (int r = 0; r < 8; ++r) {
            const int s = i0 + r + 8 * kb;
            ao[(size_t)s * EDIM + h * 64 + t * 16 + m] =
                f2bf(o[t][r] / lst[r]);
        }
    }
}

// ---------------------------------------------------------------------------
// Kernel 4: output projection  out = AO @ Wo + bo   (f32 result)
//   grid (S/64, E/64), block 128
// ---------------------------------------------------------------------------
__global__ void __launch_bounds__(128)
oproj_kernel(const u16_t* __restrict__ ao, const u16_t* __restrict__ wot,
             const float* __restrict__ bo, float* __restrict__ out)
{
    const int lane = threadIdx.x & 31;
    const int wv   = threadIdx.x >> 5;
    const int it   = blockIdx.x * 4 + wv;
    const int i0   = it * 16;
    const int n0   = blockIdx.y * 64;
    const int m    = lane & 15;
    const int kb   = lane >> 4;

    const u16_t* arow = ao + (size_t)(i0 + m) * EDIM;
    const u16_t* wrow[4];
#pragma unroll
    for (int t = 0; t < 4; ++t)
        wrow[t] = wot + (size_t)(n0 + t * 16 + m) * EDIM;

    v8f acc[4];
#pragma unroll
    for (int t = 0; t < 4; ++t) acc[t] = zero8();

    gemm_strip(arow, wrow, kb, acc);

#pragma unroll
    for (int t = 0; t < 4; ++t) {
        const float bcol = bo[n0 + t * 16 + m];
#pragma unroll
        for (int r = 0; r < 8; ++r) {
            const int s = i0 + r + 8 * kb;
            out[(size_t)s * EDIM + n0 + t * 16 + m] = acc[t][r] + bcol;
        }
    }
}

// ---------------------------------------------------------------------------
// Host launcher
// ---------------------------------------------------------------------------
extern "C" void kernel_launch(void* const* d_in, const int* in_sizes, int n_in,
                              void* d_out, int out_size, void* d_ws,
                              size_t ws_size, hipStream_t stream)
{
    const float* x   = (const float*)d_in[0];
    // d_in[1] = mask (unused: causality applied analytically)
    const float* Wq  = (const float*)d_in[2];
    const float* bq  = (const float*)d_in[3];
    const float* Wk  = (const float*)d_in[4];
    const float* bk  = (const float*)d_in[5];
    const float* Wv  = (const float*)d_in[6];
    const float* bv  = (const float*)d_in[7];
    const float* Wo  = (const float*)d_in[8];
    const float* bo  = (const float*)d_in[9];
    const float* rpe = (const float*)d_in[10];
    float* out = (float*)d_out;

    char* w = (char*)d_ws;
    size_t off = 0;
    auto alloc = [&](size_t bytes) -> void* {
        void* p = w + off;
        off += (bytes + 255) & ~(size_t)255;
        return p;
    };
    u16_t* xb   = (u16_t*)alloc((size_t)SDIM * EDIM * 2);           // x  bf16
    u16_t* wqt  = (u16_t*)alloc((size_t)EDIM * EDIM * 2);           // Wq^T
    u16_t* wkt  = (u16_t*)alloc((size_t)EDIM * EDIM * 2);           // Wk^T
    u16_t* wvt  = (u16_t*)alloc((size_t)EDIM * EDIM * 2);           // Wv^T
    u16_t* wot  = (u16_t*)alloc((size_t)EDIM * EDIM * 2);           // Wo^T
    u16_t* relb = (u16_t*)alloc((size_t)(2 * SDIM - 1) * DDIM * 2); // rel bf16
    u16_t* qh   = (u16_t*)alloc((size_t)HN * SDIM * DDIM * 2);      // Q rotated
    u16_t* kh   = (u16_t*)alloc((size_t)HN * SDIM * DDIM * 2);      // K rot*sc
    u16_t* vt   = (u16_t*)alloc((size_t)HN * DDIM * SDIM * 2);      // V^T
    u16_t* ao   = (u16_t*)alloc((size_t)SDIM * EDIM * 2);           // attn out
    (void)ws_size; (void)n_in; (void)in_sizes; (void)out_size;

    // --- prep: bf16 conversions + weight transposes ---
    int n = SDIM * EDIM;
    cvt_bf16_kernel<<<(n + 255) / 256, 256, 0, stream>>>(x, xb, n);
    n = (2 * SDIM - 1) * DDIM;
    cvt_bf16_kernel<<<(n + 255) / 256, 256, 0, stream>>>(rpe, relb, n);
    n = EDIM * EDIM;
    transpose_cvt_kernel<<<(n + 255) / 256, 256, 0, stream>>>(Wq, wqt);
    transpose_cvt_kernel<<<(n + 255) / 256, 256, 0, stream>>>(Wk, wkt);
    transpose_cvt_kernel<<<(n + 255) / 256, 256, 0, stream>>>(Wv, wvt);
    transpose_cvt_kernel<<<(n + 255) / 256, 256, 0, stream>>>(Wo, wot);

    // --- fused QKV projection + rotary ---
    qkv_kernel<<<dim3(SDIM / 64, HN, 3), 128, 0, stream>>>(
        xb, wqt, wkt, wvt, bq, bk, bv, qh, kh, vt);

    // --- flash attention with relative-position bias ---
    attn_kernel<<<dim3(SDIM / 16, HN), 32, 0, stream>>>(qh, kh, vt, relb, ao);

    // --- output projection ---
    oproj_kernel<<<dim3(SDIM / 64, EDIM / 64), 128, 0, stream>>>(
        ao, wot, bo, out);
}
